// MultiHeadLatentAttention_73280732005044
// MI455X (gfx1250) — matrix operand order repaired and checked by reading, not compiled, not run on hardware
//
#include <hip/hip_runtime.h>
#include <hip/hip_bf16.h>
#include <stdint.h>
#include <stddef.h>

// ---------------- problem constants ----------------
#define D_MODEL 1024
#define N_HEADS 8
#define D_KH    128     // per-head content dim
#define D_C     32      // latent dim
#define D_HR    32      // rope dim
#define DQK     160     // D_KH + D_HR
#define BATCH   4
#define SEQ     2048
#define NTOK    (BATCH*SEQ)                 // 8192
#define UPCOLS  (N_HEADS*DQK + N_HEADS*D_KH + N_HEADS*D_KH)  // 3328

// attention tiling
#define BM 64           // queries per workgroup (16 per wave)
#define BN 64           // keys per inner iteration
#define KSTRIDE 168     // LDS row stride for K tile (128 + 32 + 8 pad) elems
#define VSTRIDE 72      // LDS row stride for V^T tile (64 + 8 pad) elems
#define NKB (SEQ/BN)    // 32 key blocks

typedef __bf16 bf16_t;
typedef __attribute__((ext_vector_type(16))) __bf16 v16bf;
typedef __attribute__((ext_vector_type(8)))  __bf16 v8bf;
typedef __attribute__((ext_vector_type(8)))  float  v8f;
typedef __attribute__((ext_vector_type(4)))  unsigned u32x4;
typedef __attribute__((ext_vector_type(8)))  int      i32x8;
typedef __attribute__((ext_vector_type(4)))  int      i32x4;

#define HAVE_TDM 0
#if defined(__has_builtin)
#if __has_builtin(__builtin_amdgcn_tensor_load_to_lds)
#undef HAVE_TDM
#define HAVE_TDM 1
#endif
#endif

// ---------------- device helpers ----------------
__device__ __forceinline__ v16bf cat8(v8bf lo, v8bf hi) {
    v16bf r;
#pragma unroll
    for (int i = 0; i < 8; ++i) { r[i] = lo[i]; r[i + 8] = hi[i]; }
    return r;
}

// A-operand fragment (16x32 bf16): lane = row, 16 contiguous K at [h*16,+16)
__device__ __forceinline__ v16bf load_afrag(const bf16_t* row_chunk, int hh) {
    v8bf lo = *reinterpret_cast<const v8bf*>(row_chunk + hh * 16);
    v8bf hi = *reinterpret_cast<const v8bf*>(row_chunk + hh * 16 + 8);
    return cat8(lo, hi);
}

// B-operand fragment (32x16 bf16): lane = column, 16 contiguous K at [h*16,+16)
__device__ __forceinline__ v16bf load_bfrag(const bf16_t* col_chunk, int hh) {
    v8bf lo = *reinterpret_cast<const v8bf*>(col_chunk + hh * 16);
    v8bf hi = *reinterpret_cast<const v8bf*>(col_chunk + hh * 16 + 8);
    return cat8(lo, hi);
}

__device__ __forceinline__ v8f wmma_bf16(v16bf a, v16bf b, v8f c) {
    return __builtin_amdgcn_wmma_f32_16x16x32_bf16(
        false, a, false, b, (short)0, c, false, false);
}

__device__ __forceinline__ v8f zero8() {
    v8f z = {0.f, 0.f, 0.f, 0.f, 0.f, 0.f, 0.f, 0.f};
    return z;
}

__device__ __forceinline__ unsigned pack2bf(float a, float b) {
    union { bf16_t h[2]; unsigned u; } t;
    t.h[0] = (bf16_t)a; t.h[1] = (bf16_t)b;
    return t.u;
}

__device__ __forceinline__ v16bf frag_from_dwords(const unsigned* d) {
    union { v16bf v; unsigned u[8]; } t;
#pragma unroll
    for (int i = 0; i < 8; ++i) t.u[i] = d[i];
    return t.v;
}

// CDNA5 async memory->LDS copy (GVS addressing: SGPR64 base + VGPR32 offset),
// 16B per lane, tracked by ASYNCcnt. VDST = per-lane LDS byte address.
__device__ __forceinline__ void async_ld_b128(unsigned lds_off, unsigned voff, const void* base) {
    asm volatile("global_load_async_to_lds_b128 %0, %1, %2"
                 :: "v"(lds_off), "v"(voff), "s"((unsigned long long)(uintptr_t)base)
                 : "memory");
}
__device__ __forceinline__ void wait_async0() {
    asm volatile("s_wait_asynccnt 0x0" ::: "memory");
}
__device__ __forceinline__ void wait_tensor0() {
#if HAVE_TDM
    __builtin_amdgcn_s_wait_tensorcnt((short)0);
#endif
}

// TDM: one-instruction DMA of the V^T tile (128 rows x 64 bf16, row stride SEQ)
// into LDS with per-row padding to VSTRIDE. D# per CDNA5 ISA.
// This toolchain's builtin takes 6 args: (g0, g1, g2, g3, extra_v8i, cpol).
__device__ __forceinline__ void tdm_load_vtile(const bf16_t* gsrc, unsigned lds_addr) {
#if HAVE_TDM
    unsigned long long ga = (unsigned long long)(uintptr_t)gsrc;
    u32x4 g0;
    g0[0] = 1u;                                    // count=1, user mode, no gather
    g0[1] = lds_addr;                              // LDS byte address
    g0[2] = (unsigned)(ga & 0xFFFFFFFFu);          // global_addr[31:0]  (D# bits 95:64)
    g0[3] = (unsigned)((ga >> 32) & 0x1FFFFFFu)    // global_addr[56:32] (D# bits 120:96)
          | (2u << 30);                            // type=2 ("image")
    i32x8 g1;
    // data_size=1 (2B), pad_enable, pad_interval=4 (32 DWORDs = 128B row),
    // pad_amount=3 (4 DWORDs = 16B pad -> VSTRIDE*2 = 144B row pitch)
    g1[0] = (int)((1u << 16) | (1u << 20) | (4u << 22) | (3u << 25));
    g1[1] = (int)((unsigned)SEQ << 16);            // tensor_dim0[15:0] (D# bits 63:48)
    g1[2] = (int)(128u << 16);                     // tensor_dim1[15:0] (D# bits 95:80)
    g1[3] = (int)(64u << 16);                      // tile_dim0 = 64  (D# bits 127:112)
    g1[4] = 128;                                   // tile_dim1 = 128 (D# bits 143:128)
    g1[5] = SEQ;                                   // tensor_dim0_stride lo32 (bits 207:160)
    g1[6] = 0;
    g1[7] = 0;
    i32x4 z4 = {0, 0, 0, 0};                       // groups 2/3 unused (2D tensor)
    i32x8 z8 = {0, 0, 0, 0, 0, 0, 0, 0};
    __builtin_amdgcn_tensor_load_to_lds(g0, g1, z4, z4, z8, 0);
#else
    (void)gsrc; (void)lds_addr;
#endif
}

// ---------------- kernel 0: f32 -> bf16 convert ----------------
__global__ void k_cvt_bf16(const float* __restrict__ x, bf16_t* __restrict__ y, int n) {
    int i = blockIdx.x * blockDim.x + threadIdx.x;
    if (i < n) y[i] = (bf16_t)x[i];
}

// ---------------- kernel 1a: pack down-proj weights (transposed) ----------------
__global__ void k_pack_down(const float* __restrict__ W_DQ, const float* __restrict__ b_DQ,
                            const float* __restrict__ W_DKV, const float* __restrict__ b_DKV,
                            const float* __restrict__ W_KR, const float* __restrict__ b_KR,
                            bf16_t* __restrict__ WdT, float* __restrict__ bdown) {
    int idx = blockIdx.x * blockDim.x + threadIdx.x;
    if (idx < 96 * D_MODEL) {
        int n = idx / D_MODEL, k = idx % D_MODEL;
        float v = (n < 32) ? W_DQ[k * 32 + n]
                 : (n < 64) ? W_DKV[k * 32 + (n - 32)]
                            : W_KR[k * 32 + (n - 64)];
        WdT[idx] = (bf16_t)v;
    }
    if (idx < 96) {
        bdown[idx] = (idx < 32) ? b_DQ[idx] : (idx < 64) ? b_DKV[idx - 32] : b_KR[idx - 64];
    }
}

// ---------------- kernel 1b: pack up-proj weights (transposed, fused Q/K/V) ----------------
__global__ void k_pack_up(const float* __restrict__ W_UQ, const float* __restrict__ b_UQ,
                          const float* __restrict__ W_QR, const float* __restrict__ b_QR,
                          const float* __restrict__ W_UK, const float* __restrict__ b_UK,
                          const float* __restrict__ W_UV, const float* __restrict__ b_UV,
                          bf16_t* __restrict__ WuT, float* __restrict__ bup) {
    int idx = blockIdx.x * blockDim.x + threadIdx.x;
    if (idx < UPCOLS * D_C) {
        int r = idx / D_C, k = idx % D_C;
        float v;
        if (r < 1280) {
            int head = r / DQK, j = r % DQK;
            v = (j < D_KH) ? W_UQ[k * 1024 + head * D_KH + j]
                           : W_QR[k * 256 + head * D_HR + (j - D_KH)];
        } else if (r < 2304) {
            v = W_UK[k * 1024 + (r - 1280)];
        } else {
            v = W_UV[k * 1024 + (r - 2304)];
        }
        WuT[idx] = (bf16_t)v;
    }
    if (idx < UPCOLS) {
        int r = idx;
        float bv;
        if (r < 1280) {
            int head = r / DQK, j = r % DQK;
            bv = (j < D_KH) ? b_UQ[head * D_KH + j] : b_QR[head * D_HR + (j - D_KH)];
        } else if (r < 2304) bv = b_UK[r - 1280];
        else                 bv = b_UV[r - 2304];
        bup[r] = bv;
    }
}

// ---------------- kernel 1c: pack W_O transposed ----------------
__global__ void k_pack_wo(const float* __restrict__ W_O, bf16_t* __restrict__ WoT) {
    int idx = blockIdx.x * blockDim.x + threadIdx.x;
    if (idx < D_MODEL * D_MODEL) {
        int n = idx / D_MODEL, k = idx % D_MODEL;
        WoT[idx] = (bf16_t)W_O[k * D_MODEL + n];
    }
}

// ---------------- kernel 2: down projection GEMM [8192,1024]x[1024,96] ----------------
__global__ __launch_bounds__(128) void k_down(const bf16_t* __restrict__ hbf,
                                              const bf16_t* __restrict__ WdT,
                                              const float* __restrict__ bdown,
                                              bf16_t* __restrict__ cq,
                                              bf16_t* __restrict__ ckv,
                                              bf16_t* __restrict__ kr) {
    int wave = threadIdx.x >> 5, lane = threadIdx.x & 31;
    int n16 = lane & 15, hh = lane >> 4;
    int t0 = blockIdx.x * 64 + wave * 16;
    const bf16_t* Arow = hbf + (size_t)(t0 + n16) * D_MODEL;

    v8f C[6];
#pragma unroll
    for (int nt = 0; nt < 6; ++nt) C[nt] = zero8();

    v16bf a = load_afrag(Arow, hh);
    v16bf b[6];
#pragma unroll
    for (int nt = 0; nt < 6; ++nt)
        b[nt] = load_bfrag(WdT + (size_t)(nt * 16 + n16) * D_MODEL, hh);

    for (int kc = 0; kc < D_MODEL / 32; ++kc) {
        v16bf an = a;
        v16bf bn[6];
#pragma unroll
        for (int nt = 0; nt < 6; ++nt) bn[nt] = b[nt];
        if (kc + 1 < D_MODEL / 32) {
            an = load_afrag(Arow + (kc + 1) * 32, hh);
#pragma unroll
            for (int nt = 0; nt < 6; ++nt)
                bn[nt] = load_bfrag(WdT + (size_t)(nt * 16 + n16) * D_MODEL + (kc + 1) * 32, hh);
        }
#pragma unroll
        for (int nt = 0; nt < 6; ++nt) C[nt] = wmma_bf16(a, b[nt], C[nt]);
        a = an;
#pragma unroll
        for (int nt = 0; nt < 6; ++nt) b[nt] = bn[nt];
    }
#pragma unroll
    for (int nt = 0; nt < 6; ++nt) {
        int col = nt * 16 + n16;
        float bias = bdown[col];
#pragma unroll
        for (int r = 0; r < 8; ++r) {
            int token = t0 + hh * 8 + r;
            bf16_t o = (bf16_t)(C[nt][r] + bias);
            if (col < 32)       cq[(size_t)token * 32 + col] = o;
            else if (col < 64)  ckv[(size_t)token * 32 + (col - 32)] = o;
            else                kr[(size_t)token * 32 + (col - 64)] = o;
        }
    }
}

// ---------------- kernel 3: up projection (K=32) -> Q, K, V^T ----------------
__global__ __launch_bounds__(128) void k_up(const bf16_t* __restrict__ cq,
                                            const bf16_t* __restrict__ ckv,
                                            const bf16_t* __restrict__ WuT,
                                            const float* __restrict__ bup,
                                            bf16_t* __restrict__ Qg,
                                            bf16_t* __restrict__ Kg,
                                            bf16_t* __restrict__ Vtg) {
    int wave = threadIdx.x >> 5, lane = threadIdx.x & 31;
    int n16 = lane & 15, hh = lane >> 4;
    int t0 = blockIdx.x * 64 + wave * 16;
    int ct0 = blockIdx.y * 16;
    bool useQ = (ct0 * 16) < 1280;

    v16bf A = useQ ? load_afrag(cq  + (size_t)(t0 + n16) * D_C, hh)
                   : load_afrag(ckv + (size_t)(t0 + n16) * D_C, hh);

    v16bf b = load_bfrag(WuT + (size_t)(ct0 * 16 + n16) * D_C, hh);
    for (int i = 0; i < 16; ++i) {
        int ct = ct0 + i;
        v16bf bn = b;
        if (i + 1 < 16)
            bn = load_bfrag(WuT + (size_t)((ct + 1) * 16 + n16) * D_C, hh);
        v8f c = wmma_bf16(A, b, zero8());
        int col = ct * 16 + n16;
        float bias = bup[col];
#pragma unroll
        for (int r = 0; r < 8; ++r) {
            int token = t0 + hh * 8 + r;
            int bb = token >> 11, ss = token & (SEQ - 1);
            bf16_t o = (bf16_t)(c[r] + bias);
            if (col < 1280) {
                int head = col / DQK, j = col % DQK;
                Qg[((size_t)(bb * N_HEADS + head) * SEQ + ss) * DQK + j] = o;
            } else if (col < 2304) {
                int cc = col - 1280; int head = cc >> 7, j = cc & 127;
                Kg[((size_t)(bb * N_HEADS + head) * SEQ + ss) * D_KH + j] = o;
            } else {
                int cc = col - 2304; int head = cc >> 7, d = cc & 127;
                Vtg[((size_t)(bb * N_HEADS + head) * D_KH + d) * SEQ + ss] = o;
            }
        }
        b = bn;
    }
}

// ---------------- kernel 4: fused flash attention ----------------
// Double-buffered LDS tiles: K/rope rows via global_load_async_to_lds_b128
// (ASYNCcnt), V^T tile via one TDM tensor_load_to_lds per WG (TENSORcnt),
// tile kb+1 in flight while computing tile kb.
__global__ __launch_bounds__(128) void k_attn(const bf16_t* __restrict__ Qg,
                                              const bf16_t* __restrict__ Kg,
                                              const bf16_t* __restrict__ KRg,
                                              const bf16_t* __restrict__ Vtg,
                                              bf16_t* __restrict__ AO) {
    __shared__ bf16_t Ksh[2][BN * KSTRIDE];
    __shared__ bf16_t Vsh[2][D_KH * VSTRIDE];

    int wg = blockIdx.x;
    int qblk = wg % (SEQ / BM);
    int bh = wg / (SEQ / BM);
    int b = bh / N_HEADS;
    int h = bh % N_HEADS;

    int tid = threadIdx.x;
    int wave = tid >> 5, lane = tid & 31;
    int q16 = lane & 15, hh = lane >> 4;
    int q0 = qblk * BM + wave * 16;

    auto stage = [&](int kb, int buf) {
        const bf16_t* srcK = Kg + ((size_t)bh * SEQ + kb * BN) * D_KH;
        bf16_t* dK = &Ksh[buf][0];
#pragma unroll
        for (int it = tid; it < BN * D_KH / 8; it += 128) {
            int r = it / (D_KH / 8), cc = it % (D_KH / 8);
            async_ld_b128((unsigned)(uintptr_t)&dK[r * KSTRIDE + cc * 8],
                          (unsigned)((r * D_KH + cc * 8) * 2), srcK);
        }
        const bf16_t* srcR = KRg + ((size_t)b * SEQ + kb * BN) * D_HR;
#pragma unroll
        for (int it = tid; it < BN * D_HR / 8; it += 128) {
            int r = it / (D_HR / 8), cc = it % (D_HR / 8);
            async_ld_b128((unsigned)(uintptr_t)&dK[r * KSTRIDE + D_KH + cc * 8],
                          (unsigned)((r * D_HR + cc * 8) * 2), srcR);
        }
        const bf16_t* srcV = Vtg + (size_t)bh * D_KH * SEQ + (size_t)kb * BN;
        bf16_t* dV = &Vsh[buf][0];
#if HAVE_TDM
        if (wave == 0)
            tdm_load_vtile(srcV, (unsigned)(uintptr_t)dV);
#else
#pragma unroll
        for (int it = tid; it < D_KH * BN / 8; it += 128) {
            int r = it / (BN / 8), cc = it % (BN / 8);
            async_ld_b128((unsigned)(uintptr_t)&dV[r * VSTRIDE + cc * 8],
                          (unsigned)((r * SEQ + cc * 8) * 2), srcV);
        }
#endif
    };

    // Q^T B-operand fragments, resident for the whole loop
    const bf16_t* Qrow = Qg + ((size_t)bh * SEQ + q0 + q16) * DQK;
    v16bf Qf[5];
#pragma unroll
    for (int c = 0; c < 5; ++c) Qf[c] = load_bfrag(Qrow + c * 32, hh);

    v8f Oacc[8];
#pragma unroll
    for (int d = 0; d < 8; ++d) Oacc[d] = zero8();
    float m = -3.0e38f, l = 0.f;
    const float scale = 0.07905694150420949f;  // 1/sqrt(160)

    stage(0, 0);

    for (int kb = 0; kb < NKB; ++kb) {
        int cur = kb & 1;
        wait_async0();          // this wave's async chunks of tile kb have landed
        wait_tensor0();         // wave0: V^T TDM for tile kb complete
        __syncthreads();        // everyone's copies landed; buf[cur^1] free to refill
        if (kb + 1 < NKB) stage(kb + 1, cur ^ 1);

        const bf16_t* Kb = &Ksh[cur][0];
        const bf16_t* Vb = &Vsh[cur][0];

        // ---- S^T = K . Q^T : 4 key-subtiles of 16
        v8f Sf[4];
#pragma unroll
        for (int ks = 0; ks < 4; ++ks) {
            const bf16_t* krow = Kb + (ks * 16 + q16) * KSTRIDE;
            v16bf a[5];
#pragma unroll
            for (int cc = 0; cc < 5; ++cc) a[cc] = load_afrag(krow + cc * 32, hh);
            v8f c = zero8();
#pragma unroll
            for (int cc = 0; cc < 5; ++cc) c = wmma_bf16(a[cc], Qf[cc], c);
            Sf[ks] = c;
        }

        // ---- online softmax (lane holds 32 of this query's 64 scores)
        float mloc = -3.0e38f;
#pragma unroll
        for (int ks = 0; ks < 4; ++ks)
#pragma unroll
            for (int r = 0; r < 8; ++r) {
                Sf[ks][r] *= scale;
                mloc = fmaxf(mloc, Sf[ks][r]);
            }
        mloc = fmaxf(mloc, __shfl_xor(mloc, 16, 32));
        float mnew = fmaxf(m, mloc);
        float alpha = __expf(m - mnew);
        float ls = 0.f;
        unsigned pk[4][4];
#pragma unroll
        for (int ks = 0; ks < 4; ++ks) {
            float p[8];
#pragma unroll
            for (int r = 0; r < 8; ++r) { p[r] = __expf(Sf[ks][r] - mnew); ls += p[r]; }
#pragma unroll
            for (int d = 0; d < 4; ++d) pk[ks][d] = pack2bf(p[2 * d], p[2 * d + 1]);
        }
        m = mnew;
        l = l * alpha + ls;
#pragma unroll
        for (int d = 0; d < 8; ++d)
#pragma unroll
            for (int r = 0; r < 8; ++r) Oacc[d][r] *= alpha;

        // ---- build P^T B-operand fragments via partner half-swap
        unsigned dw0[8], dw1[8];
#pragma unroll
        for (int d = 0; d < 4; ++d) {
            unsigned o0 = (unsigned)__shfl_xor((int)pk[0][d], 16, 32);
            unsigned o1 = (unsigned)__shfl_xor((int)pk[1][d], 16, 32);
            unsigned o2 = (unsigned)__shfl_xor((int)pk[2][d], 16, 32);
            unsigned o3 = (unsigned)__shfl_xor((int)pk[3][d], 16, 32);
            dw0[d]     = hh ? o1       : pk[0][d];
            dw0[d + 4] = hh ? pk[1][d] : o0;
            dw1[d]     = hh ? o3       : pk[2][d];
            dw1[d + 4] = hh ? pk[3][d] : o2;
        }
        v16bf Pf0 = frag_from_dwords(dw0);
        v16bf Pf1 = frag_from_dwords(dw1);

        // ---- O^T += V^T . P^T : 8 dim-subtiles x 2 key-chunks
#pragma unroll
        for (int ds = 0; ds < 8; ++ds) {
            const bf16_t* vrow = Vb + (ds * 16 + q16) * VSTRIDE;
            v16bf a0 = load_afrag(vrow, hh);
            v16bf a1 = load_afrag(vrow + 32, hh);
            Oacc[ds] = wmma_bf16(a0, Pf0, Oacc[ds]);
            Oacc[ds] = wmma_bf16(a1, Pf1, Oacc[ds]);
        }
    }

    float lt = l + __shfl_xor(l, 16, 32);
    float inv = 1.0f / lt;
    size_t orow = ((size_t)b * SEQ + q0 + q16) * (N_HEADS * D_KH) + (size_t)h * D_KH;
#pragma unroll
    for (int ds = 0; ds < 8; ++ds)
#pragma unroll
        for (int r = 0; r < 8; ++r) {
            int dim = ds * 16 + hh * 8 + r;
            AO[orow + dim] = (bf16_t)(Oacc[ds][r] * inv);
        }
}

// ---------------- kernel 5: output projection (software-pipelined) ----------------
__global__ __launch_bounds__(128) void k_out(const bf16_t* __restrict__ AO,
                                             const bf16_t* __restrict__ WoT,
                                             const float* __restrict__ bO,
                                             float* __restrict__ out) {
    int wave = threadIdx.x >> 5, lane = threadIdx.x & 31;
    int n16 = lane & 15, hh = lane >> 4;
    int t0 = blockIdx.x * 64 + wave * 16;
    int c0 = blockIdx.y * 64;
    const bf16_t* Arow = AO + (size_t)(t0 + n16) * D_MODEL;

    v8f C[4];
#pragma unroll
    for (int nt = 0; nt < 4; ++nt) C[nt] = zero8();

    v16bf a = load_afrag(Arow, hh);
    v16bf b[4];
#pragma unroll
    for (int nt = 0; nt < 4; ++nt)
        b[nt] = load_bfrag(WoT + (size_t)(c0 + nt * 16 + n16) * D_MODEL, hh);

    for (int kc = 0; kc < D_MODEL / 32; ++kc) {
        __builtin_prefetch(Arow + (kc + 8) * 32, 0, 0);   // global_prefetch_b8
        v16bf an = a;
        v16bf bn[4];
#pragma unroll
        for (int nt = 0; nt < 4; ++nt) bn[nt] = b[nt];
        if (kc + 1 < D_MODEL / 32) {
            an = load_afrag(Arow + (kc + 1) * 32, hh);
#pragma unroll
            for (int nt = 0; nt < 4; ++nt)
                bn[nt] = load_bfrag(WoT + (size_t)(c0 + nt * 16 + n16) * D_MODEL + (kc + 1) * 32, hh);
        }
#pragma unroll
        for (int nt = 0; nt < 4; ++nt) C[nt] = wmma_bf16(a, b[nt], C[nt]);
        a = an;
#pragma unroll
        for (int nt = 0; nt < 4; ++nt) b[nt] = bn[nt];
    }
#pragma unroll
    for (int nt = 0; nt < 4; ++nt) {
        int col = c0 + nt * 16 + n16;
        float bias = bO[col];
#pragma unroll
        for (int r = 0; r < 8; ++r) {
            int token = t0 + hh * 8 + r;
            out[(size_t)token * D_MODEL + col] = C[nt][r] + bias;
        }
    }
}

// ---------------- launcher ----------------
extern "C" void kernel_launch(void* const* d_in, const int* in_sizes, int n_in,
                              void* d_out, int out_size, void* d_ws, size_t ws_size,
                              hipStream_t stream) {
    const float* h_t   = (const float*)d_in[0];
    const float* W_DQ  = (const float*)d_in[1];
    const float* b_DQ  = (const float*)d_in[2];
    const float* W_UQ  = (const float*)d_in[3];
    const float* b_UQ  = (const float*)d_in[4];
    const float* W_DKV = (const float*)d_in[5];
    const float* b_DKV = (const float*)d_in[6];
    const float* W_UK  = (const float*)d_in[7];
    const float* b_UK  = (const float*)d_in[8];
    const float* W_UV  = (const float*)d_in[9];
    const float* b_UV  = (const float*)d_in[10];
    const float* W_QR  = (const float*)d_in[11];
    const float* b_QR  = (const float*)d_in[12];
    const float* W_KR  = (const float*)d_in[13];
    const float* b_KR  = (const float*)d_in[14];
    const float* W_O   = (const float*)d_in[15];
    const float* b_O   = (const float*)d_in[16];
    float* out = (float*)d_out;

    char* ws = (char*)d_ws;
    size_t off = 0;
    auto carve = [&](size_t bytes) { void* p = ws + off; off += (bytes + 255) & ~(size_t)255; return p; };
    bf16_t* hbf  = (bf16_t*)carve((size_t)NTOK * D_MODEL * 2);
    bf16_t* WdT  = (bf16_t*)carve((size_t)96 * D_MODEL * 2);
    float*  bdn  = (float*) carve(96 * 4);
    bf16_t* WuT  = (bf16_t*)carve((size_t)UPCOLS * D_C * 2);
    float*  bup  = (float*) carve(UPCOLS * 4);
    bf16_t* WoT  = (bf16_t*)carve((size_t)D_MODEL * D_MODEL * 2);
    bf16_t* cq   = (bf16_t*)carve((size_t)NTOK * D_C * 2);
    bf16_t* ckv  = (bf16_t*)carve((size_t)NTOK * D_C * 2);
    bf16_t* kr   = (bf16_t*)carve((size_t)NTOK * D_HR * 2);
    bf16_t* Qg   = (bf16_t*)carve((size_t)BATCH * N_HEADS * SEQ * DQK * 2);
    bf16_t* Kg   = (bf16_t*)carve((size_t)BATCH * N_HEADS * SEQ * D_KH * 2);
    bf16_t* Vtg  = (bf16_t*)carve((size_t)BATCH * N_HEADS * D_KH * SEQ * 2);
    bf16_t* AO   = (bf16_t*)carve((size_t)NTOK * (N_HEADS * D_KH) * 2);
    (void)ws_size; (void)in_sizes; (void)n_in; (void)out_size;

    {
        int n = NTOK * D_MODEL;
        k_cvt_bf16<<<(n + 255) / 256, 256, 0, stream>>>(h_t, hbf, n);
        k_pack_down<<<(96 * D_MODEL + 255) / 256, 256, 0, stream>>>(
            W_DQ, b_DQ, W_DKV, b_DKV, W_KR, b_KR, WdT, bdn);
        k_pack_up<<<(UPCOLS * D_C + 255) / 256, 256, 0, stream>>>(
            W_UQ, b_UQ, W_QR, b_QR, W_UK, b_UK, W_UV, b_UV, WuT, bup);
        k_pack_wo<<<(D_MODEL * D_MODEL + 255) / 256, 256, 0, stream>>>(W_O, WoT);
    }
    k_down<<<NTOK / 64, 128, 0, stream>>>(hbf, WdT, bdn, cq, ckv, kr);
    k_up<<<dim3(NTOK / 64, UPCOLS / 16 / 16), 128, 0, stream>>>(cq, ckv, WuT, bup, Qg, Kg, Vtg);
    k_attn<<<BATCH * N_HEADS * (SEQ / BM), 128, 0, stream>>>(Qg, Kg, kr, Vtg, AO);
    k_out<<<dim3(NTOK / 64, D_MODEL / 64), 128, 0, stream>>>(AO, WoT, b_O, out);
}
